// MaskSelfAttention_23845658428239
// MI455X (gfx1250) — matrix operand order repaired and checked by reading, not compiled, not run on hardware
//
#include <hip/hip_runtime.h>
#include <math.h>

// ---------------- problem constants ----------------
#define Bb 8
#define Nn 1024
#define Dd 1024
#define Hh 16
#define DHd 64
#define SCALE 0.125f
#define NEGV -1e9f

// GEMM tiling
#define BM 128
#define BN 64
#define BK 32
#define LDP (BK + 8)   // padded LDS row stride (40 bf16 = 80B, 16B aligned, bank-friendly)

typedef __bf16  bf16;
typedef __bf16  v4bf  __attribute__((ext_vector_type(4)));
typedef __bf16  v8bf  __attribute__((ext_vector_type(8)));
typedef __bf16  v16bf __attribute__((ext_vector_type(16)));
typedef float   v4f   __attribute__((ext_vector_type(4)));
typedef float   v8f   __attribute__((ext_vector_type(8)));

static __device__ __forceinline__ bf16 f2bf(float f) { return (bf16)f; }   // native v_cvt RNE
static __device__ __forceinline__ unsigned short f2bfu(float f) {
    return __builtin_bit_cast(unsigned short, (bf16)f);
}
static __device__ __forceinline__ v16bf joinv(v8bf lo, v8bf hi) {
    return __builtin_shufflevector(lo, hi, 0,1,2,3,4,5,6,7,8,9,10,11,12,13,14,15);
}
static __device__ __forceinline__ v8f v8f_zero() {
    v8f r;
#pragma unroll
    for (int i = 0; i < 8; ++i) r[i] = 0.0f;
    return r;
}
static __device__ __forceinline__ v8f wmma_bf16(v16bf a, v16bf b, v8f c) {
    return __builtin_amdgcn_wmma_f32_16x16x32_bf16(false, a, false, b, (short)0, c, false, false);
}

// A fragment (16x32 bf16): lane=row (mod 16); elems = K[half*16 .. +15] contiguous
static __device__ __forceinline__ v16bf load_afrag(const bf16* row_base, int half) {
    const bf16* p = row_base + half * 16;
    return joinv(*(const v8bf*)p, *(const v8bf*)(p + 8));
}
// B fragment (32x16 bf16): lane=col (mod 16); elems = K[half*16 .. +15] contiguous
static __device__ __forceinline__ v16bf load_bfrag(const bf16* col_base, int half) {
    const bf16* p = col_base + half * 16;
    return joinv(*(const v8bf*)p, *(const v8bf*)(p + 8));
}

// =====================================================================
// Kernel 1: qkv = x @ w_qkv   (M=8192, N=3072, K=1024), fp32 in, bf16 out,
// scattered to q[B,H,N,DH], k[B,H,N,DH], vT[B,H,DH,N]
// =====================================================================
__global__ void __launch_bounds__(256, 2)
k_qkv_gemm(const float* __restrict__ x, const float* __restrict__ w,
           unsigned short* __restrict__ qb, unsigned short* __restrict__ kb,
           unsigned short* __restrict__ vtb) {
    __shared__ bf16 lA[BM][LDP];
    __shared__ bf16 lBt[BN][LDP];   // B stored transposed: [n][k]

    const int t    = threadIdx.x;
    const int wave = t >> 5, lane = t & 31;
    const int wm   = wave & 3, wn = wave >> 2;          // 4x2 wave grid -> 128x64
    const int m0   = blockIdx.y * BM;
    const int n0   = blockIdx.x * BN;
    const int half = lane >> 4, lr = lane & 15;

    v8f acc[2][2];
#pragma unroll
    for (int i = 0; i < 2; ++i)
#pragma unroll
        for (int j = 0; j < 2; ++j) acc[i][j] = v8f_zero();

    for (int k0 = 0; k0 < Dd; k0 += BK) {
        if (k0 + BK < Dd) {  // prefetch next K tiles (near-cache)
            __builtin_prefetch(&x[(size_t)(m0 + (t >> 1)) * Dd + k0 + BK + (t & 1) * 16], 0, 3);
            __builtin_prefetch(&w[(size_t)(k0 + BK + (t >> 3)) * (3 * Dd) + n0 + (t & 7) * 8], 0, 3);
        }
#pragma unroll
        for (int i = 0; i < 4; ++i) {               // A: 128x32 f32 -> bf16 (float4 loads)
            int flat = t + i * 256;                 // 1024 float4 chunks
            int r = flat >> 3, c4 = (flat & 7) * 4;
            v4f xv = *(const v4f*)&x[(size_t)(m0 + r) * Dd + k0 + c4];
            v4bf bv;
#pragma unroll
            for (int j = 0; j < 4; ++j) bv[j] = f2bf(xv[j]);
            *(v4bf*)&lA[r][c4] = bv;
        }
#pragma unroll
        for (int i = 0; i < 2; ++i) {               // B: 32x64 f32 -> bf16, transposed
            int flat = t + i * 256;                 // 512 float4 chunks
            int kr = flat >> 4, c4 = (flat & 15) * 4;
            v4f wv = *(const v4f*)&w[(size_t)(k0 + kr) * (3 * Dd) + n0 + c4];
#pragma unroll
            for (int j = 0; j < 4; ++j) lBt[c4 + j][kr] = f2bf(wv[j]);
        }
        __syncthreads();

        v16bf af[2], bfr[2];
#pragma unroll
        for (int tm = 0; tm < 2; ++tm)
            af[tm] = load_afrag(&lA[wm * 32 + tm * 16 + lr][0], half);
#pragma unroll
        for (int tn = 0; tn < 2; ++tn)
            bfr[tn] = load_bfrag(&lBt[wn * 32 + tn * 16 + lr][0], half);
#pragma unroll
        for (int tm = 0; tm < 2; ++tm)
#pragma unroll
            for (int tn = 0; tn < 2; ++tn)
                acc[tm][tn] = wmma_bf16(af[tm], bfr[tn], acc[tm][tn]);
        __syncthreads();
    }

    // epilogue: segment (q/k/v) is uniform per block since BN | 1024
    auto scatter_nat = [&](unsigned short* __restrict__ dst, int cb) {
#pragma unroll
        for (int tm = 0; tm < 2; ++tm)
#pragma unroll
            for (int tn = 0; tn < 2; ++tn) {
                int col = cb + wn * 32 + tn * 16 + lr;
                int h = col >> 6, dh = col & 63;
#pragma unroll
                for (int r = 0; r < 8; ++r) {
                    int row = m0 + wm * 32 + tm * 16 + half * 8 + r;
                    int b = row >> 10, n = row & (Nn - 1);
                    dst[(((size_t)(b * Hh + h)) * Nn + n) * DHd + dh] = f2bfu(acc[tm][tn][r]);
                }
            }
    };
    if (n0 < Dd) {
        scatter_nat(qb, n0);
    } else if (n0 < 2 * Dd) {
        scatter_nat(kb, n0 - Dd);
    } else {
        int cb = n0 - 2 * Dd;
#pragma unroll
        for (int tm = 0; tm < 2; ++tm)
#pragma unroll
            for (int tn = 0; tn < 2; ++tn) {
                int col = cb + wn * 32 + tn * 16 + lr;
                int h = col >> 6, dh = col & 63;
#pragma unroll
                for (int r = 0; r < 8; ++r) {
                    int row = m0 + wm * 32 + tm * 16 + half * 8 + r;
                    int b = row >> 10, n = row & (Nn - 1);
                    vtb[(((size_t)(b * Hh + h)) * DHd + dh) * Nn + n] = f2bfu(acc[tm][tn][r]);
                }
            }
    }
}

// =====================================================================
// Kernel 2: per (b,h,16-query-tile): scores -> mask -> softmax -> attn out
// (f32) + probs@V -> out_heads (bf16).  128 threads = 4 waves.
// Dynamic LDS: f32 scores[16][1024] | bf16 probs[16][1024] | red[16][8] | stat[16]
// =====================================================================
__global__ void __launch_bounds__(128, 1)
k_attn(const unsigned short* __restrict__ qb_u, const unsigned short* __restrict__ kb_u,
       const unsigned short* __restrict__ vtb_u, const int* __restrict__ mask,
       float* __restrict__ attn_out, unsigned short* __restrict__ oh) {
    extern __shared__ char smem[];
    float* sc   = (float*)smem;                         // 16*1024 f32  (64KB)
    bf16*  pb   = (bf16*)(smem + 16 * 1024 * 4);        // 16*1024 bf16 (32KB)
    float* red  = (float*)(smem + 16 * 1024 * 6);       // 16*8
    float* stat = red + 128;                            // 16

    const bf16* qb  = (const bf16*)qb_u;
    const bf16* kb  = (const bf16*)kb_u;
    const bf16* vtb = (const bf16*)vtb_u;

    const int t = threadIdx.x, wave = t >> 5, lane = t & 31;
    const int bh = blockIdx.x >> 6;          // b*H + h
    const int qt = blockIdx.x & 63;
    const int b  = bh >> 4, h = bh & 15;
    const int q0 = qt * 16;
    const int half = lane >> 4, lr = lane & 15;
    const size_t bhbase = (size_t)bh * Nn * DHd;

    // Q fragments for the two K=32 steps of DH=64
    v16bf qf[2];
    {
        const bf16* qrow = qb + bhbase + (size_t)(q0 + lr) * DHd;
#pragma unroll
        for (int s = 0; s < 2; ++s) qf[s] = load_afrag(qrow + s * 32, half);
    }

    // ---- scores: each wave covers 256 keys (16 key tiles of 16) ----
    for (int i = 0; i < 16; ++i) {
        int kt   = wave * 16 + i;
        int kcol = kt * 16 + lr;                        // key column for this lane
        const bf16* krow = kb + bhbase + (size_t)kcol * DHd;
        v8f acc = v8f_zero();
#pragma unroll
        for (int s = 0; s < 2; ++s)
            acc = wmma_bf16(qf[s], load_bfrag(krow + s * 32, half), acc);
        int mk = mask[b * Nn + kcol];
#pragma unroll
        for (int r = 0; r < 8; ++r) {
            float sv = (mk == 0) ? NEGV : acc[r] * SCALE;
            sc[(half * 8 + r) * Nn + kcol] = sv;
        }
    }
    __syncthreads();

    // ---- softmax over 1024 keys; thread t: row = t>>3, float4 chunks ----
    {
        int row = t >> 3, seg = t & 7;
        float* srow = sc + row * Nn;
        float m = -3.4e38f;
        for (int i = 0; i < 32; ++i) {
            v4f s4 = *(const v4f*)&srow[(seg + i * 8) * 4];
            m = fmaxf(m, fmaxf(fmaxf(s4[0], s4[1]), fmaxf(s4[2], s4[3])));
        }
        red[row * 8 + seg] = m;
        __syncthreads();
        if (seg == 0) {
            float mm = red[row * 8];
#pragma unroll
            for (int j = 1; j < 8; ++j) mm = fmaxf(mm, red[row * 8 + j]);
            stat[row] = mm;
        }
        __syncthreads();
        float rmax = stat[row], sum = 0.f;
        for (int i = 0; i < 32; ++i) {
            int cb = (seg + i * 8) * 4;
            v4f s4 = *(const v4f*)&srow[cb];
#pragma unroll
            for (int j = 0; j < 4; ++j) { float e = __expf(s4[j] - rmax); s4[j] = e; sum += e; }
            *(v4f*)&srow[cb] = s4;
        }
        red[row * 8 + seg] = sum;
        __syncthreads();
        if (seg == 0) {
            float ss = 0.f;
#pragma unroll
            for (int j = 0; j < 8; ++j) ss += red[row * 8 + j];
            stat[row] = 1.0f / ss;
        }
        __syncthreads();
        float inv = stat[row];
        float* arow = attn_out + ((size_t)bh * Nn + q0 + row) * Nn;
        bf16*  prow = pb + row * Nn;
        for (int i = 0; i < 32; ++i) {
            int cb = (seg + i * 8) * 4;
            v4f s4 = *(const v4f*)&srow[cb];
            v4f p4; v4bf pbv;
#pragma unroll
            for (int j = 0; j < 4; ++j) { p4[j] = s4[j] * inv; pbv[j] = f2bf(p4[j]); }
            *(v4f*)&arow[cb] = p4;                      // 16B global store
            *(v4bf*)&prow[cb] = pbv;                    // 8B LDS store
        }
    }
    __syncthreads();

    // ---- out_tile[16 x 64] = probs[16x1024] @ V[1024x64]; wave -> 16 dh cols ----
    v8f oacc = v8f_zero();
    const bf16* vbase = vtb + (size_t)bh * DHd * Nn + (size_t)(wave * 16 + lr) * Nn;
    for (int ks = 0; ks < 32; ++ks) {
        v16bf af  = load_afrag(pb + (size_t)lr * Nn + ks * 32, half);
        v16bf bfr = load_bfrag(vbase + ks * 32, half);
        oacc = wmma_bf16(af, bfr, oacc);
    }
    {
        int col = h * DHd + wave * 16 + lr;
#pragma unroll
        for (int r = 0; r < 8; ++r) {
            int row = q0 + half * 8 + r;
            oh[((size_t)(b * Nn + row)) * Dd + col] = f2bfu(oacc[r]);
        }
    }
}

// =====================================================================
// Kernel 3: out = out_heads(bf16) @ w_out + b_out   (M=8192, N=1024, K=1024)
// A tile staged via GLOBAL_LOAD_ASYNC_TO_LDS_B128 (ASYNCcnt path)
// =====================================================================
__global__ void __launch_bounds__(256, 2)
k_outproj(const unsigned short* __restrict__ Au, const float* __restrict__ w,
          const float* __restrict__ bias, float* __restrict__ out) {
    __shared__ bf16 lA[BM][LDP];
    __shared__ bf16 lBt[BN][LDP];

    const int t    = threadIdx.x;
    const int wave = t >> 5, lane = t & 31;
    const int wm   = wave & 3, wn = wave >> 2;
    const int m0   = blockIdx.y * BM;
    const int n0   = blockIdx.x * BN;
    const int half = lane >> 4, lr = lane & 15;

    v8f acc[2][2];
#pragma unroll
    for (int i = 0; i < 2; ++i)
#pragma unroll
        for (int j = 0; j < 2; ++j) acc[i][j] = v8f_zero();

    for (int k0 = 0; k0 < Dd; k0 += BK) {
        if (k0 + BK < Dd)
            __builtin_prefetch(&w[(size_t)(k0 + BK + (t >> 3)) * Dd + n0 + (t & 7) * 8], 0, 3);

        // ---- A tile (128x32 bf16 = 8KB): async copy global -> LDS, 2 x 16B/lane ----
#pragma unroll
        for (int i = 0; i < 2; ++i) {
            int flat = t + i * 256;                 // 512 chunks of 8 bf16
            int r = flat >> 2, c8 = (flat & 3) * 8;
            unsigned long long ga =
                (unsigned long long)(size_t)(Au + (size_t)(m0 + r) * Dd + k0 + c8);
            unsigned loff = (unsigned)(size_t)&lA[r][c8];   // LDS byte offset (addr[31:0])
            asm volatile("global_load_async_to_lds_b128 %0, %1, off"
                         :: "v"(loff), "v"(ga) : "memory");
        }
        // ---- B tile: f32 -> bf16, transposed, via VGPRs ----
#pragma unroll
        for (int i = 0; i < 2; ++i) {
            int flat = t + i * 256;
            int kr = flat >> 4, c4 = (flat & 15) * 4;
            v4f wv = *(const v4f*)&w[(size_t)(k0 + kr) * Dd + n0 + c4];
#pragma unroll
            for (int j = 0; j < 4; ++j) lBt[c4 + j][kr] = f2bf(wv[j]);
        }
        asm volatile("s_wait_asynccnt 0" ::: "memory");
        __syncthreads();

        v16bf af[2], bfr[2];
#pragma unroll
        for (int tm = 0; tm < 2; ++tm)
            af[tm] = load_afrag(&lA[wm * 32 + tm * 16 + lr][0], half);
#pragma unroll
        for (int tn = 0; tn < 2; ++tn)
            bfr[tn] = load_bfrag(&lBt[wn * 32 + tn * 16 + lr][0], half);
#pragma unroll
        for (int tm = 0; tm < 2; ++tm)
#pragma unroll
            for (int tn = 0; tn < 2; ++tn)
                acc[tm][tn] = wmma_bf16(af[tm], bfr[tn], acc[tm][tn]);
        __syncthreads();
    }

#pragma unroll
    for (int tm = 0; tm < 2; ++tm) {
#pragma unroll
        for (int tn = 0; tn < 2; ++tn) {
            int col = n0 + wn * 32 + tn * 16 + lr;
            float bv = bias[col];
#pragma unroll
            for (int r = 0; r < 8; ++r) {
                int row = m0 + wm * 32 + tm * 16 + half * 8 + r;
                out[(size_t)row * Dd + col] = acc[tm][tn][r] + bv;
            }
        }
    }
}

// =====================================================================
extern "C" void kernel_launch(void* const* d_in, const int* in_sizes, int n_in,
                              void* d_out, int out_size, void* d_ws, size_t ws_size,
                              hipStream_t stream) {
    (void)in_sizes; (void)n_in; (void)out_size; (void)ws_size;
    const float* x     = (const float*)d_in[0];
    const int*   mask  = (const int*)d_in[1];
    const float* w_qkv = (const float*)d_in[2];
    const float* w_out = (const float*)d_in[3];
    const float* b_out = (const float*)d_in[4];

    float* outp  = (float*)d_out;                               // [B,N,D]
    float* attnp = outp + (size_t)Bb * Nn * Dd;                 // [B,H,N,N]

    const size_t SEG = (size_t)Bb * Hh * Nn * DHd;              // 8,388,608 elems
    unsigned short* qb  = (unsigned short*)d_ws;
    unsigned short* kb  = qb  + SEG;
    unsigned short* vtb = kb  + SEG;
    unsigned short* oh  = vtb + SEG;                            // [B*N, D] bf16

    dim3 g1((3 * Dd) / BN, (Bb * Nn) / BM);                     // 48 x 64
    k_qkv_gemm<<<g1, 256, 0, stream>>>(x, w_qkv, qb, kb, vtb);

    size_t smem = 16 * 1024 * 4 + 16 * 1024 * 2 + 16 * 8 * 4 + 16 * 4;  // 98,880 B
    k_attn<<<dim3(Bb * Hh * (Nn / 16)), 128, smem, stream>>>(qb, kb, vtb, mask, attnp, oh);

    dim3 g3(Dd / BN, (Bb * Nn) / BM);                           // 16 x 64
    k_outproj<<<g3, 256, 0, stream>>>(oh, w_out, b_out, outp);
}
